// Discriminator_6279242187454
// MI455X (gfx1250) — matrix operand order repaired and checked by reading, not compiled, not run on hardware
//
#include <hip/hip_runtime.h>
#include <hip/hip_bf16.h>
#include <math.h>
#include <stdint.h>

// ---------------------------------------------------------------------------
// CDNA5 (gfx1250) wave32 WMMA building blocks
// ---------------------------------------------------------------------------
typedef __attribute__((ext_vector_type(16))) _Float16 v16h;
typedef __attribute__((ext_vector_type(8)))  _Float16 v8h;
typedef __attribute__((ext_vector_type(8)))  float    v8f;
typedef __attribute__((ext_vector_type(4)))  float    v4f;

// One 16x16x32 WMMA step with vectorized (b128) LDS fragment loads.
//   As : 16x32 f16 tile, K-major rows (stride lda_s halves)
//   BsT: 16(n) x 32(k) f16 tile, N-major (stride ldb_s halves) == B transposed
// ISA 7.12.2 fragment maps:
//   A: lane row lr=lane&15, halves e -> k = (e>>3)*16 + hi*8 + (e&7)
//   B: lane col lr, halves e -> k = (e>>3)*16 + hi*8 + (e&7)
//   C: element r -> (m = hi*8 + r, n = lr)
__device__ __forceinline__ v8f wmma_nt(const _Float16* As, int lda_s,
                                       const _Float16* BsT, int ldb_s, v8f acc) {
  const int lane = threadIdx.x & 31;
  const int lr = lane & 15;
  const int hi = lane >> 4;
  union { v16h v; v8h h[2]; } ua, ub;
  const _Float16* ap = As + lr * lda_s + (hi << 3);
  ua.h[0] = *(const v8h*)ap;          // k = hi*8 + 0..7
  ua.h[1] = *(const v8h*)(ap + 16);   // k = 16 + hi*8 + 0..7
  const _Float16* bp = BsT + lr * ldb_s + (hi << 3);
  ub.h[0] = *(const v8h*)bp;          // k = hi*8 + 0..7
  ub.h[1] = *(const v8h*)(bp + 16);   // k = 16 + hi*8 + 0..7
  return __builtin_amdgcn_wmma_f32_16x16x32_f16(false, ua.v, false, ub.v,
                                                (short)0, acc, false, false);
}

__device__ __forceinline__ float sigmoidf_(float x) { return 1.f / (1.f + expf(-x)); }

// ---------------------------------------------------------------------------
// Implicit-GEMM convolution (NHWC f32 src, OIHW f32 weights, NHWC f32 dst).
// K ordering: k = (kh*KW + kw)*Cin + cin (cin-minor). For power-of-two Cin
// (>=8, always a multiple of 32 total K here) a thread's 8 consecutive k's are
// one spatial tap + 8 consecutive channels -> coalesced float4 NHWC loads and
// shift/mask decode. Out-of-range M/N rows are CLAMPED (not branched): their
// garbage only reaches C elements that are never stored.
// Block tile 64x64, K-step 32, double-buffered LDS, 8 waves = 4(M) x 2(N),
// each wave a 16x32 C strip (two accumulators, A fragment reused).
// ---------------------------------------------------------------------------
template <int KH, int KW>
__global__ void __launch_bounds__(256)
conv2d_wmma(const float* __restrict__ src, const float* __restrict__ wgt,
            float* __restrict__ dst,
            int Bn, int H, int W, int Cin,
            int OH, int OW, int Cout,
            int stride, int pad) {
  const int M = Bn * OH * OW;
  const int Kreal = Cin * KH * KW;
  const int K32 = (Kreal + 31) & ~31;
  const int T = K32 >> 5;          // number of 32-wide K tiles
  const int OHW = OH * OW;
  const bool cpow2 = (Cin >= 8) && ((Cin & (Cin - 1)) == 0);
  const int cshift = 31 - __builtin_clz(Cin > 0 ? Cin : 1);
  const int cmask = Cin - 1;

  __shared__ _Float16 As[2][64][40];    // double-buffered 64 x 32k (padded)
  __shared__ _Float16 BsT[2][64][40];   // double-buffered 64 x 32k, N-major

  const int tid = threadIdx.x;
  const int m0 = blockIdx.y * 64;
  const int n0 = blockIdx.x * 64;

  // A loader: thread -> (row ra = tid>>2, 8 k's at (tid&3)*8); invariant m-decode.
  const int ra = tid >> 2;
  const int ca0 = (tid & 3) * 8;
  const int m = m0 + ra;
  int mc = m < M ? m : (M - 1);           // clamped row (garbage rows unused)
  const int ib = mc / OHW;
  {
  }
  const int rrm = mc - ib * OHW;
  const int oh = rrm / OW;
  const int ow = rrm - oh * OW;
  const int ih0 = oh * stride - pad;
  const int iw0 = ow * stride - pad;
  const float* srcb = src + (size_t)ib * H * W * Cin;

  // B loader: thread -> (n-row nb = tid>>2, 8 k's at (tid&3)*8)
  const int nb = tid >> 2;
  const int kb0 = (tid & 3) * 8;
  int n_b = n0 + nb;
  if (n_b >= Cout) n_b = Cout - 1;        // clamped column

  const int wv = tid >> 5;
  const int wr = wv & 3;        // wave m-row 0..3
  const int wc = wv >> 2;       // wave n-col 0..1
  v8f acc0 = {}, acc1 = {};

  v8h arg, brg;
  auto load_tile = [&](int kk) {
    // ---- A (im2col gather) ----
    {
      const int k0 = kk + ca0;
      _Float16 t[8];
      if (cpow2) {
        // cpow2 here implies Kreal % 32 == 0 -> k0+7 < Kreal always
        const int sp = k0 >> cshift;
        const int cin0 = k0 & cmask;
        const int kh = sp / KW;
        const int kw = sp - kh * KW;
        const int ih = ih0 + kh, iw = iw0 + kw;
        if ((unsigned)ih < (unsigned)H && (unsigned)iw < (unsigned)W) {
          const float* p = srcb + ((size_t)ih * W + iw) * Cin + cin0;
          v4f x0 = *(const v4f*)p;
          v4f x1 = *(const v4f*)(p + 4);
#pragma unroll
          for (int i = 0; i < 4; ++i) { t[i] = (_Float16)x0[i]; t[4 + i] = (_Float16)x1[i]; }
        } else {
#pragma unroll
          for (int i = 0; i < 8; ++i) t[i] = (_Float16)0.f;
        }
      } else {
        // generic path (first conv: Cin==3), per-element guarded decode
#pragma unroll
        for (int i = 0; i < 8; ++i) {
          const int k = k0 + i;
          float v = 0.f;
          if (k < Kreal) {
            const int sp = k / Cin;
            const int cin = k - sp * Cin;
            const int kh = sp / KW;
            const int kw = sp - kh * KW;
            const int ih = ih0 + kh, iw = iw0 + kw;
            if ((unsigned)ih < (unsigned)H && (unsigned)iw < (unsigned)W)
              v = srcb[((size_t)ih * W + iw) * Cin + cin];
          }
          t[i] = (_Float16)v;
        }
      }
      arg = *(const v8h*)t;
    }
    // ---- B (OIHW weights, N-major, same K ordering) ----
    {
      const int k0 = kk + kb0;
      _Float16 t[8];
      if (cpow2) {
        const int sp = k0 >> cshift;
        const int cin0 = k0 & cmask;
        const int kh = sp / KW;
        const int kw = sp - kh * KW;
        const float* wp = wgt + (((size_t)n_b * Cin + cin0) * KH + kh) * KW + kw;
#pragma unroll
        for (int i = 0; i < 8; ++i) t[i] = (_Float16)wp[(size_t)i * (KH * KW)];
      } else {
#pragma unroll
        for (int i = 0; i < 8; ++i) {
          const int k = k0 + i;
          float v = 0.f;
          if (k < Kreal) {
            const int sp = k / Cin;
            const int cin = k - sp * Cin;
            const int kh = sp / KW;
            const int kw = sp - kh * KW;
            v = wgt[(((size_t)n_b * Cin + cin) * KH + kh) * KW + kw];
          }
          t[i] = (_Float16)v;
        }
      }
      brg = *(const v8h*)t;
    }
  };

  // prologue: stage tile 0
  load_tile(0);
  *(v8h*)&As[0][ra][ca0] = arg;
  *(v8h*)&BsT[0][nb][kb0] = brg;
  __syncthreads();

  for (int t = 0; t < T; ++t) {
    const int cur = t & 1;
    const int nxt = cur ^ 1;
    if (t + 1 < T) load_tile((t + 1) << 5);     // global prefetch overlaps WMMA
    const _Float16* Ab = &As[cur][wr * 16][0];
    acc0 = wmma_nt(Ab, 40, &BsT[cur][wc * 32][0], 40, acc0);
    acc1 = wmma_nt(Ab, 40, &BsT[cur][wc * 32 + 16][0], 40, acc1);
    if (t + 1 < T) {
      *(v8h*)&As[nxt][ra][ca0] = arg;
      *(v8h*)&BsT[nxt][nb][kb0] = brg;
    }
    __syncthreads();
  }

  const int lane = tid & 31;
  const int lr = lane & 15;
  const int hi = lane >> 4;
#pragma unroll
  for (int r = 0; r < 8; ++r) {
    const int mm = m0 + wr * 16 + hi * 8 + r;
    if (mm < M) {
      const int nn0 = n0 + wc * 32 + lr;
      const int nn1 = nn0 + 16;
      if (nn0 < Cout) dst[(size_t)mm * Cout + nn0] = acc0[r];
      if (nn1 < Cout) dst[(size_t)mm * Cout + nn1] = acc1[r];
    }
  }
}

// ---------------------------------------------------------------------------
// Generic linear GEMM: C[M,N] = A[M,K] * B[N,K]^T + bias[N]  (f32 in/out,
// f16 staged in LDS; f32 WMMA accumulate). Same 64x64 double-buffered tiling.
// Main K loop is unconditional (clamped rows, vec4 loads); single guarded
// tail tile zero-fills K padding.
// ---------------------------------------------------------------------------
__global__ void __launch_bounds__(256)
gemm_bias_wmma(const float* __restrict__ A, int lda,
               const float* __restrict__ Bw, int ldb,
               const float* __restrict__ bias,
               float* __restrict__ C, int M, int N, int K) {
  const int K32 = (K + 31) & ~31;
  const int T = K32 >> 5;
  __shared__ _Float16 As[2][64][40];
  __shared__ _Float16 BsT[2][64][40];

  const int tid = threadIdx.x;
  const int m0 = blockIdx.y * 64;
  const int n0 = blockIdx.x * 64;

  const int ra = tid >> 2;
  const int ca0 = (tid & 3) * 8;
  int mc = m0 + ra; if (mc >= M) mc = M - 1;      // clamped row
  const int nb = tid >> 2;
  const int kb0 = (tid & 3) * 8;
  int nc = n0 + nb; if (nc >= N) nc = N - 1;      // clamped column

  // uniform fast-path selectors
  const bool a_vec4 = ((lda & 3) == 0) && ((((uintptr_t)A) & 15) == 0);
  const bool b_vec4 = ((ldb & 3) == 0) && ((((uintptr_t)Bw) & 15) == 0);

  const int wv = tid >> 5;
  const int wr = wv & 3;
  const int wc = wv >> 2;
  v8f acc0 = {}, acc1 = {};

  v8h arg, brg;
  auto load_tile = [&](int kk) {
    const bool guard = (kk + 32) > K;            // uniform: only last tile
    {
      const float* ap = A + (size_t)mc * lda + kk + ca0;
      _Float16 t[8];
      if (!guard && a_vec4) {
        v4f x0 = *(const v4f*)ap;
        v4f x1 = *(const v4f*)(ap + 4);
#pragma unroll
        for (int i = 0; i < 4; ++i) { t[i] = (_Float16)x0[i]; t[4 + i] = (_Float16)x1[i]; }
      } else {
#pragma unroll
        for (int i = 0; i < 8; ++i) {
          const int k = kk + ca0 + i;
          t[i] = (_Float16)((k < K) ? ap[i] : 0.f);
        }
      }
      arg = *(const v8h*)t;
    }
    {
      const float* bp = Bw + (size_t)nc * ldb + kk + kb0;
      _Float16 t[8];
      if (!guard && b_vec4) {
        v4f x0 = *(const v4f*)bp;
        v4f x1 = *(const v4f*)(bp + 4);
#pragma unroll
        for (int i = 0; i < 4; ++i) { t[i] = (_Float16)x0[i]; t[4 + i] = (_Float16)x1[i]; }
      } else {
#pragma unroll
        for (int i = 0; i < 8; ++i) {
          const int k = kk + kb0 + i;
          t[i] = (_Float16)((k < K) ? bp[i] : 0.f);
        }
      }
      brg = *(const v8h*)t;
    }
  };

  load_tile(0);
  *(v8h*)&As[0][ra][ca0] = arg;
  *(v8h*)&BsT[0][nb][kb0] = brg;
  __syncthreads();

  for (int t = 0; t < T; ++t) {
    const int cur = t & 1;
    const int nxt = cur ^ 1;
    if (t + 1 < T) load_tile((t + 1) << 5);
    const _Float16* Ab = &As[cur][wr * 16][0];
    acc0 = wmma_nt(Ab, 40, &BsT[cur][wc * 32][0], 40, acc0);
    acc1 = wmma_nt(Ab, 40, &BsT[cur][wc * 32 + 16][0], 40, acc1);
    if (t + 1 < T) {
      *(v8h*)&As[nxt][ra][ca0] = arg;
      *(v8h*)&BsT[nxt][nb][kb0] = brg;
    }
    __syncthreads();
  }

  const int lane = tid & 31;
  const int lr = lane & 15;
  const int hi = lane >> 4;
#pragma unroll
  for (int r = 0; r < 8; ++r) {
    const int mm = m0 + wr * 16 + hi * 8 + r;
    if (mm < M) {
      const int nn0 = n0 + wc * 32 + lr;
      const int nn1 = nn0 + 16;
      if (nn0 < N) C[(size_t)mm * N + nn0] = acc0[r] + (bias ? bias[nn0] : 0.f);
      if (nn1 < N) C[(size_t)mm * N + nn1] = acc1[r] + (bias ? bias[nn1] : 0.f);
    }
  }
}

// ---------------------------------------------------------------------------
// Elementwise / reduction helper kernels
// ---------------------------------------------------------------------------
__global__ void nchw_to_nhwc(const float* __restrict__ src, float* __restrict__ dst,
                             int Bn, int C, int H, int W) {
  size_t idx = (size_t)blockIdx.x * blockDim.x + threadIdx.x;
  size_t total = (size_t)Bn * C * H * W;
  if (idx >= total) return;
  int c = (int)(idx % C); size_t t = idx / C;
  int w = (int)(t % W); t /= W;
  int h = (int)(t % H);
  int b = (int)(t / H);
  dst[idx] = src[(((size_t)b * C + c) * H + h) * W + w];
}

__global__ void zero_kernel(float* p, size_t n) {
  size_t i = (size_t)blockIdx.x * blockDim.x + threadIdx.x;
  if (i < n) p[i] = 0.f;
}

__global__ void bn_stats_kernel(const float* __restrict__ x, float* __restrict__ stats,
                                int MC, int C) {
  const int c = blockIdx.x;
  __shared__ float rs[256], rq[256];
  float s = 0.f, q = 0.f;
  for (int i = threadIdx.x; i < MC; i += blockDim.x) {
    float v = x[(size_t)i * C + c];
    s += v; q += v * v;
  }
  rs[threadIdx.x] = s; rq[threadIdx.x] = q; __syncthreads();
  for (int o = 128; o > 0; o >>= 1) {
    if (threadIdx.x < o) { rs[threadIdx.x] += rs[threadIdx.x + o]; rq[threadIdx.x] += rq[threadIdx.x + o]; }
    __syncthreads();
  }
  if (threadIdx.x == 0) {
    float m = rs[0] / MC;
    float var = rq[0] / MC - m * m;
    stats[c] = m;
    stats[C + c] = rsqrtf(fmaxf(var, 0.f) + 1e-5f);
  }
}

__global__ void bn_lrelu_kernel(float* __restrict__ x, const float* __restrict__ stats,
                                const float* __restrict__ g, const float* __restrict__ b,
                                size_t total, int C) {
  size_t idx = (size_t)blockIdx.x * blockDim.x + threadIdx.x;
  if (idx >= total) return;
  float v = x[idx];
  if (g) {
    int c = (int)(idx % C);
    v = g[c] * (v - stats[c]) * stats[C + c] + b[c];
  }
  x[idx] = v >= 0.f ? v : 0.2f * v;
}

// GRU gate nonlinearity + state update + masked output.
__global__ void gru_gate_kernel(const float* __restrict__ gx_base, int gx_bstride,
                                const float* __restrict__ gh,
                                const float* __restrict__ h_in, float* __restrict__ h_out,
                                float* __restrict__ out_mat,
                                const int* __restrict__ len_text,
                                int t_mask, int Hd, int Bn) {
  int idx = blockIdx.x * blockDim.x + threadIdx.x;
  if (idx >= Bn * Hd) return;
  int b = idx / Hd, j = idx - b * Hd;
  const float* gxr = gx_base + (size_t)b * gx_bstride;
  const float* ghr = gh + (size_t)b * 3 * Hd;
  float r = sigmoidf_(gxr[j] + ghr[j]);
  float z = sigmoidf_(gxr[Hd + j] + ghr[Hd + j]);
  float n = tanhf(gxr[2 * Hd + j] + r * ghr[2 * Hd + j]);
  float h = h_in[(size_t)b * Hd + j];
  float hn = (1.f - z) * n + z * h;
  h_out[(size_t)b * Hd + j] = hn;
  float msk = (t_mask < len_text[b]) ? 1.f : 0.f;
  out_mat[(size_t)b * Hd + j] = hn * msk;
}

// words[b][s][h] = 0.5*(f_mat[s][b][h] + b_mat[s][b][h]),
// with b_mat[0]=v[0], b_mat[s]=v[S-s] for s>=1.
__global__ void words_kernel(const float* __restrict__ f_mat, const float* __restrict__ v_mat,
                             float* __restrict__ words, int S, int Bn, int Hd) {
  size_t idx = (size_t)blockIdx.x * blockDim.x + threadIdx.x;
  size_t total = (size_t)Bn * S * Hd;
  if (idx >= total) return;
  int h = (int)(idx % Hd); size_t t = idx / Hd;
  int s = (int)(t % S);
  int b = (int)(t / S);
  int sv = (s == 0) ? 0 : (S - s);
  float f = f_mat[((size_t)s * Bn + b) * Hd + h];
  float v = v_mat[((size_t)sv * Bn + b) * Hd + h];
  words[idx] = 0.5f * (f + v);
}

__global__ void avg_kernel(const float* __restrict__ words, const int* __restrict__ len_text,
                           float* __restrict__ avg, int S, int Hd, int Bn) {
  int idx = blockIdx.x * blockDim.x + threadIdx.x;
  if (idx >= Bn * Hd) return;
  int b = idx / Hd, h = idx - b * Hd;
  float s = 0.f;
  for (int t = 0; t < S; ++t) s += words[((size_t)b * S + t) * Hd + h];
  int cnt = len_text[b]; if (cnt > S) cnt = S;
  avg[(size_t)b * Hd + h] = s / (float)cnt;
}

__global__ void alphas_kernel(const float* __restrict__ words, const float* __restrict__ avg,
                              float* __restrict__ alphas, int S, int Hd) {
  const int b = blockIdx.x;
  __shared__ float red[256];
  __shared__ float u[32];
  for (int s = 0; s < S; ++s) {
    float p = 0.f;
    for (int h = threadIdx.x; h < Hd; h += blockDim.x)
      p += words[((size_t)b * S + s) * Hd + h] * avg[(size_t)b * Hd + h];
    red[threadIdx.x] = p; __syncthreads();
    for (int o = 128; o > 0; o >>= 1) {
      if (threadIdx.x < o) red[threadIdx.x] += red[threadIdx.x + o];
      __syncthreads();
    }
    if (threadIdx.x == 0) u[s] = red[0];
    __syncthreads();
  }
  if (threadIdx.x == 0) {
    float mx = -3.4e38f;
    for (int s = 0; s < S; ++s) mx = fmaxf(mx, u[s]);
    float sm = 0.f;
    for (int s = 0; s < S; ++s) { float e = expf(u[s] - mx); u[s] = e; sm += e; }
    for (int s = 0; s < S; ++s) alphas[(size_t)b * S + s] = u[s] / sm;
  }
}

__global__ void betas_logits_kernel(const float* __restrict__ words, const float* __restrict__ Wm,
                                    const float* __restrict__ bv, float* __restrict__ out,
                                    int BS, int Hd) {
  int idx = blockIdx.x * blockDim.x + threadIdx.x;
  if (idx >= BS * 3) return;
  int bs = idx / 3, j = idx - bs * 3;
  float s = bv[j];
  for (int h = 0; h < Hd; ++h) s += words[(size_t)bs * Hd + h] * Wm[(size_t)j * Hd + h];
  out[idx] = s;
}

// softmax over batch axis (axis 0 of [B,S,3]) for each (s,j)
__global__ void betas_softmax_kernel(const float* __restrict__ logits, float* __restrict__ sm,
                                     int Bn, int S) {
  int sj = blockIdx.x;
  int s = sj / 3, j = sj - s * 3;
  if (threadIdx.x != 0) return;
  float mx = -3.4e38f;
  for (int b = 0; b < Bn; ++b) mx = fmaxf(mx, logits[((size_t)b * S + s) * 3 + j]);
  float sum = 0.f;
  for (int b = 0; b < Bn; ++b) sum += expf(logits[((size_t)b * S + s) * 3 + j] - mx);
  for (int b = 0; b < Bn; ++b)
    sm[((size_t)b * S + s) * 3 + j] = expf(logits[((size_t)b * S + s) * 3 + j] - mx) / sum;
}

__global__ void feat_kernel(const float* __restrict__ G, float* __restrict__ feat,
                            int Bn, int HW, int C) {
  int idx = blockIdx.x * blockDim.x + threadIdx.x;
  if (idx >= Bn * C) return;
  int b = idx / C, c = idx - b * C;
  float s = 0.f;
  for (int i = 0; i < HW; ++i) s += G[((size_t)b * HW + i) * C + c];
  feat[idx] = s / (float)HW;
}

__global__ void scale_f_kernel(const float* __restrict__ Wbm, int No,
                               const float* __restrict__ feat, int C,
                               const float* __restrict__ betas_sm, int j,
                               float* __restrict__ total, int S, int acc) {
  const int bs = blockIdx.x;
  const int b = bs / S;
  __shared__ float red[256];
  float p = 0.f;
  for (int c = threadIdx.x; c < C; c += blockDim.x)
    p += Wbm[(size_t)bs * No + c] * feat[(size_t)b * C + c];
  red[threadIdx.x] = p; __syncthreads();
  for (int o = 128; o > 0; o >>= 1) {
    if (threadIdx.x < o) red[threadIdx.x] += red[threadIdx.x + o];
    __syncthreads();
  }
  if (threadIdx.x == 0) {
    float f = sigmoidf_(red[0] + Wbm[(size_t)bs * No + C]);
    float v = f * betas_sm[(size_t)bs * 3 + j];
    total[bs] = acc ? (total[bs] + v) : v;
  }
}

__global__ void final_kernel(const float* __restrict__ alphas, const float* __restrict__ total,
                             float* __restrict__ out, int S) {
  const int b = blockIdx.x;
  if (threadIdx.x != 0) return;
  float s = 0.f;
  for (int t = 0; t < S; ++t)
    s += alphas[(size_t)b * S + t] * logf(total[(size_t)b * S + t]);
  out[b] = expf(s);
}

// ---------------------------------------------------------------------------
// Launch sequence
// ---------------------------------------------------------------------------
extern "C" void kernel_launch(void* const* d_in, const int* in_sizes, int n_in,
                              void* d_out, int out_size, void* d_ws, size_t ws_size,
                              hipStream_t stream) {
  (void)in_sizes; (void)n_in; (void)out_size; (void)ws_size;
  const float* image    = (const float*)d_in[0];   // (64,3,128,128)
  const float* text     = (const float*)d_in[1];   // (64,24,300)
  const int*   len_text = (const int*)  d_in[2];   // (64,)
  const float* c3_w1 = (const float*)d_in[3];
  const float* c3_w2 = (const float*)d_in[4];
  const float* bn3_2_g = (const float*)d_in[5];
  const float* bn3_2_b = (const float*)d_in[6];
  const float* c3_w3 = (const float*)d_in[7];
  const float* bn3_3_g = (const float*)d_in[8];
  const float* bn3_3_b = (const float*)d_in[9];
  const float* c4_w = (const float*)d_in[10];
  const float* bn4_g = (const float*)d_in[11];
  const float* bn4_b = (const float*)d_in[12];
  const float* c5_w = (const float*)d_in[13];
  const float* bn5_g = (const float*)d_in[14];
  const float* bn5_b = (const float*)d_in[15];
  const float* g1_w = (const float*)d_in[16];
  const float* bn_g1_g = (const float*)d_in[17];
  const float* bn_g1_b = (const float*)d_in[18];
  const float* g2_w = (const float*)d_in[19];
  const float* bn_g2_g = (const float*)d_in[20];
  const float* bn_g2_b = (const float*)d_in[21];
  const float* g3_w = (const float*)d_in[22];
  const float* bn_g3_g = (const float*)d_in[23];
  const float* bn_g3_b = (const float*)d_in[24];
  const float* betas_W = (const float*)d_in[25];
  const float* betas_b = (const float*)d_in[26];
  const float* Wi_f = (const float*)d_in[27];
  const float* Wh_f = (const float*)d_in[28];
  const float* bi_f = (const float*)d_in[29];
  const float* bh_f = (const float*)d_in[30];
  const float* Wi_b = (const float*)d_in[31];
  const float* Wh_b = (const float*)d_in[32];
  const float* bi_b = (const float*)d_in[33];
  const float* bh_b = (const float*)d_in[34];
  const float* Wb1_W = (const float*)d_in[35];
  const float* Wb1_b = (const float*)d_in[36];
  const float* Wb2_W = (const float*)d_in[37];
  const float* Wb2_b = (const float*)d_in[38];
  float* out = (float*)d_out;

  const int Bn = 64, S = 24, Hd = 512;

  // ---- workspace carve (floats) ----
  float* ws = (float*)d_ws;
  size_t off = 0;
  auto alloc = [&](size_t n) { float* p = ws + off; off += n; return p; };
  float* img_nhwc = alloc((size_t)64 * 128 * 128 * 3);
  float* act1 = alloc((size_t)64 * 64 * 64 * 64);     // dead after conv2 -> tail arena
  float* act2 = alloc((size_t)64 * 32 * 32 * 128);
  float* img1 = alloc((size_t)64 * 16 * 16 * 256);
  float* img2 = alloc((size_t)64 * 8 * 8 * 512);
  float* img3 = alloc((size_t)64 * 4 * 4 * 512);
  float* G1 = alloc((size_t)64 * 16 * 16 * 256);
  float* G2 = alloc((size_t)64 * 8 * 8 * 512);
  float* G3 = alloc((size_t)64 * 4 * 4 * 512);
  float* stats = alloc(1024);
  // tail arena overlays act1 (16.7M floats available, ~8.2M used)
  size_t toff = 0;
  auto talloc = [&](size_t n) { float* p = act1 + toff; toff += n; return p; };
  float* gx_f = talloc((size_t)Bn * S * 1536);
  float* gx_b = talloc((size_t)Bn * S * 1536);
  float* gh   = talloc((size_t)Bn * 1536);
  float* h_a  = talloc((size_t)Bn * Hd);
  float* h_b  = talloc((size_t)Bn * Hd);
  float* f_mat = talloc((size_t)S * Bn * Hd);
  float* v_mat = talloc((size_t)S * Bn * Hd);
  float* words = talloc((size_t)Bn * S * Hd);
  float* avg   = talloc((size_t)Bn * Hd);
  float* alphas = talloc((size_t)Bn * S);
  float* blog  = talloc((size_t)Bn * S * 3);
  float* bsm   = talloc((size_t)Bn * S * 3);
  float* feat1 = talloc((size_t)Bn * 256);
  float* feat2 = talloc((size_t)Bn * 512);
  float* feat3 = talloc((size_t)Bn * 512);
  float* Wbm   = talloc((size_t)Bn * S * 513);
  float* total = talloc((size_t)Bn * S);

  auto conv = [&](const float* src, const float* w, float* dst,
                  int B_, int H_, int W_, int Cin, int Cout, int KHW_, int st, int pd) {
    int OH = (H_ + 2 * pd - KHW_) / st + 1;
    int OW = (W_ + 2 * pd - KHW_) / st + 1;
    int M = B_ * OH * OW;
    dim3 grid((Cout + 63) / 64, (M + 63) / 64);
    if (KHW_ == 4)
      conv2d_wmma<4, 4><<<grid, 256, 0, stream>>>(src, w, dst, B_, H_, W_, Cin, OH, OW,
                                                  Cout, st, pd);
    else
      conv2d_wmma<3, 3><<<grid, 256, 0, stream>>>(src, w, dst, B_, H_, W_, Cin, OH, OW,
                                                  Cout, st, pd);
  };
  auto bn_act = [&](float* x, const float* g, const float* b, int MC, int C) {
    bn_stats_kernel<<<C, 256, 0, stream>>>(x, stats, MC, C);
    size_t tot = (size_t)MC * C;
    bn_lrelu_kernel<<<(unsigned)((tot + 255) / 256), 256, 0, stream>>>(x, stats, g, b, tot, C);
  };
  auto gemm = [&](const float* A, int lda, const float* Bw, int ldb, const float* bias,
                  float* C, int M, int N, int K) {
    dim3 grid((N + 63) / 64, (M + 63) / 64);
    gemm_bias_wmma<<<grid, 256, 0, stream>>>(A, lda, Bw, ldb, bias, C, M, N, K);
  };

  // ---- image branch ----
  {
    size_t tot = (size_t)64 * 3 * 128 * 128;
    nchw_to_nhwc<<<(unsigned)((tot + 255) / 256), 256, 0, stream>>>(image, img_nhwc, 64, 3, 128, 128);
  }
  conv(img_nhwc, c3_w1, act1, 64, 128, 128, 3, 64, 4, 2, 1);           // -> (64,64,64,64)
  {
    size_t tot = (size_t)64 * 64 * 64 * 64;
    bn_lrelu_kernel<<<(unsigned)((tot + 255) / 256), 256, 0, stream>>>(act1, nullptr, nullptr, nullptr, tot, 64);
  }
  conv(act1, c3_w2, act2, 64, 64, 64, 64, 128, 4, 2, 1);               // -> (64,32,32,128)
  bn_act(act2, bn3_2_g, bn3_2_b, 64 * 32 * 32, 128);
  conv(act2, c3_w3, img1, 64, 32, 32, 128, 256, 4, 2, 1);              // -> (64,16,16,256)
  bn_act(img1, bn3_3_g, bn3_3_b, 64 * 16 * 16, 256);
  conv(img1, c4_w, img2, 64, 16, 16, 256, 512, 4, 2, 1);               // -> (64,8,8,512)
  bn_act(img2, bn4_g, bn4_b, 64 * 8 * 8, 512);
  conv(img2, c5_w, img3, 64, 8, 8, 512, 512, 4, 2, 1);                 // -> (64,4,4,512)
  bn_act(img3, bn5_g, bn5_b, 64 * 4 * 4, 512);
  conv(img1, g1_w, G1, 64, 16, 16, 256, 256, 3, 1, 1);
  bn_act(G1, bn_g1_g, bn_g1_b, 64 * 16 * 16, 256);
  conv(img2, g2_w, G2, 64, 8, 8, 512, 512, 3, 1, 1);
  bn_act(G2, bn_g2_g, bn_g2_b, 64 * 8 * 8, 512);
  conv(img3, g3_w, G3, 64, 4, 4, 512, 512, 3, 1, 1);
  bn_act(G3, bn_g3_g, bn_g3_b, 64 * 4 * 4, 512);

  // ---- text branch (act1 is dead now; tail arena overlays it) ----
  gemm(text, 300, Wi_f, 300, bi_f, gx_f, Bn * S, 1536, 300);
  gemm(text, 300, Wi_b, 300, bi_b, gx_b, Bn * S, 1536, 300);

  const int bh_blocks = (Bn * Hd + 255) / 256;
  float *hin = h_a, *hout = h_b;
  zero_kernel<<<bh_blocks, 256, 0, stream>>>(hin, (size_t)Bn * Hd);
  for (int t = 0; t < S; ++t) {
    gemm(hin, Hd, Wh_f, Hd, bh_f, gh, Bn, 1536, Hd);
    gru_gate_kernel<<<bh_blocks, 256, 0, stream>>>(gx_f + (size_t)t * 1536, S * 1536, gh,
                                                   hin, hout, f_mat + (size_t)t * Bn * Hd,
                                                   len_text, t, Hd, Bn);
    float* tmp = hin; hin = hout; hout = tmp;
  }
  zero_kernel<<<bh_blocks, 256, 0, stream>>>(hin, (size_t)Bn * Hd);
  for (int i = 0; i < S; ++i) {
    const int tx = S - 1 - i;   // xs reversed; mask index NOT reversed (reference quirk)
    gemm(hin, Hd, Wh_b, Hd, bh_b, gh, Bn, 1536, Hd);
    gru_gate_kernel<<<bh_blocks, 256, 0, stream>>>(gx_b + (size_t)tx * 1536, S * 1536, gh,
                                                   hin, hout, v_mat + (size_t)i * Bn * Hd,
                                                   len_text, i, Hd, Bn);
    float* tmp = hin; hin = hout; hout = tmp;
  }

  {
    size_t tot = (size_t)Bn * S * Hd;
    words_kernel<<<(unsigned)((tot + 255) / 256), 256, 0, stream>>>(f_mat, v_mat, words, S, Bn, Hd);
  }
  avg_kernel<<<bh_blocks, 256, 0, stream>>>(words, len_text, avg, S, Hd, Bn);
  alphas_kernel<<<Bn, 256, 0, stream>>>(words, avg, alphas, S, Hd);
  betas_logits_kernel<<<(Bn * S * 3 + 255) / 256, 256, 0, stream>>>(words, betas_W, betas_b,
                                                                    blog, Bn * S, Hd);
  betas_softmax_kernel<<<S * 3, 32, 0, stream>>>(blog, bsm, Bn, S);

  feat_kernel<<<(Bn * 256 + 255) / 256, 256, 0, stream>>>(G1, feat1, Bn, 256, 256);
  feat_kernel<<<(Bn * 512 + 255) / 256, 256, 0, stream>>>(G2, feat2, Bn, 64, 512);
  feat_kernel<<<(Bn * 512 + 255) / 256, 256, 0, stream>>>(G3, feat3, Bn, 16, 512);

  // scale 0: Wb1 (257 cols), G1 feat (256)
  gemm(words, Hd, Wb1_W, Hd, Wb1_b, Wbm, Bn * S, 257, Hd);
  scale_f_kernel<<<Bn * S, 256, 0, stream>>>(Wbm, 257, feat1, 256, bsm, 0, total, S, 0);
  // scale 1: Wb2 (513 cols), G2 feat (512)
  gemm(words, Hd, Wb2_W, Hd, Wb2_b, Wbm, Bn * S, 513, Hd);
  scale_f_kernel<<<Bn * S, 256, 0, stream>>>(Wbm, 513, feat2, 512, bsm, 1, total, S, 1);
  // scale 2: Wb2 again (reference reuses it), G3 feat (512)
  gemm(words, Hd, Wb2_W, Hd, Wb2_b, Wbm, Bn * S, 513, Hd);
  scale_f_kernel<<<Bn * S, 256, 0, stream>>>(Wbm, 513, feat3, 512, bsm, 2, total, S, 1);

  final_kernel<<<Bn, 32, 0, stream>>>(alphas, total, out, S);
}
